// CRF_21766894256321
// MI455X (gfx1250) — matrix operand order repaired and checked by reading, not compiled, not run on hardware
//
#include <hip/hip_runtime.h>
#include <hip/hip_bf16.h>
#include <cstddef>
#include <cstdint>

// CRF forward scan, B=512, S=1024, T=48 tags.
// d_in[0] = feats  [512,1024,48] f32
// d_in[1] = mask   [512,1024]    bool (1 byte/elem, jnp bool_)
// d_in[2] = trans  [48,48]       f32
// d_out   = [ sum(score), score[512] ]  (513 f32)

typedef __attribute__((ext_vector_type(16))) _Float16     v16h;
typedef __attribute__((ext_vector_type(8)))  float        v8f;
typedef __attribute__((ext_vector_type(4)))  unsigned int u32x4;
typedef __attribute__((ext_vector_type(8)))  int          i32x8;
typedef __attribute__((ext_vector_type(4)))  int          i32x4;

#define B_TOT 512
#define S_LEN 1024
#define T     48
#define TTP   52              // padded row stride of transposed transition (bank spread)
#define BPB   8               // batches per block
#define NBLK  (B_TOT / BPB)   // 64 blocks
#define TPB   (BPB * T)       // 384 threads = 12 wave32s; 1 thread per (m,j)

#define LOG2E 1.4426950408889634f
#define LN2   0.6931471805599453f

// Permuted K-position inside a 64-half row so each WMMA lane's v16h A-fragment
// is a contiguous 32B LDS read (CDNA5 16-bit A-matrix 16x32 layout, ISA 7.12.2):
//   lane<16 : A0 = K{0..7,16..23},  A1 = K{32..39,48..55}
//   lane>=16: A0 = K{8..15,24..31}, A1 = K{40..47,56..63}
__device__ __forceinline__ int aperm_pos(int k) {
    return (k & 7) + ((k >> 3) & 1) * 32 + (k >> 4) * 8;
}

__global__ __launch_bounds__(TPB, 1)
void crf_fwd(const float* __restrict__ feats,
             const unsigned char* __restrict__ mask,
             const float* __restrict__ trans,
             float* __restrict__ out,
             float* __restrict__ ws)
{
    __shared__ __attribute__((aligned(16))) float partS[BPB * T];   // forward state (f32)
    __shared__ __attribute__((aligned(16))) float PS[BPB * T];      // P[m][j] + 2*e[m][j]
    __shared__ __attribute__((aligned(16))) float eBuf[2][BPB * T]; // double-buffered emissions
    __shared__ __attribute__((aligned(8)))  unsigned char maskBuf[2][BPB];
    __shared__ __attribute__((aligned(32))) _Float16 aPerm[16 * 64];// WMMA A, K padded to 64
    __shared__ __attribute__((aligned(16))) float transT[T * TTP];  // trans transposed: [j][i]
    __shared__ float scoreSh[BPB];

    const int tid  = threadIdx.x;
    const int lane = tid & 31;
    const int wave = tid >> 5;
    const int b0   = blockIdx.x * BPB;
    const int m    = tid / T;        // batch row 0..7
    const int j    = tid - m * T;    // tag column 0..47

    // ---- init ----
    for (int o = tid; o < T * T; o += TPB) {          // transposed transition copy
        const int i = o / T, jj = o - i * T;
        transT[jj * TTP + i] = trans[o];
    }
    for (int o = tid; o < 16 * 64; o += TPB) aPerm[o] = (_Float16)0.0f;

    // part0 = emit[0] + transition[-2,:]  (row 46)
    partS[tid] = feats[((size_t)(b0 + m) * S_LEN + 0) * T + j] + trans[46 * T + j];
    // stage t=1 emissions/mask
    eBuf[1][tid] = feats[((size_t)(b0 + m) * S_LEN + 1) * T + j];
    if (tid < BPB) maskBuf[1][tid] = mask[(size_t)(b0 + tid) * S_LEN + 1];

    // Constant B-operand: E = exp(trans) in f16, resident in registers per WMMA wave.
    // CDNA5 16-bit B-matrix 32x16: lanes<16 -> K{0..7,16..23}, lanes>=16 -> K{8..15,24..31}; N = lane&15.
    v16h Bm0 = {}, Bm1 = {};
    if (wave < 3) {
        const int n  = wave * 16 + (lane & 15);
        const int hi = (lane >> 4) & 1;
        #pragma unroll
        for (int h = 0; h < 16; ++h) {
            const int k0 = (h & 7) + 8 * hi + 16 * (h >> 3);          // 0..31  (< 48 always)
            const int k1 = 32 + (h & 7) + 8 * hi + 16 * (h >> 3);     // 32..63 (>=48 -> zero pad)
            Bm0[h] = (_Float16)__expf(trans[k0 * T + n]);
            Bm1[h] = (k1 < T) ? (_Float16)__expf(trans[k1 * T + n]) : (_Float16)0.0f;
        }
    }
    __syncthreads();

    // ---- sequential scan over time ----
    #pragma unroll 1
    for (int t = 1; t < S_LEN; ++t) {
        const int cur = t & 1, nxt = cur ^ 1;

        // Phase A (all 384 threads, one (m,j) each):
        //   P = max_i(part[m][i] + trans[i][j]) via float4 LDS reads;
        //   PS = P + 2e;  a[m][j] = exp2((part - P - e) * log2e)  -> f16 A-operand
        float pmax = -3.402823466e38f;
        {
            const float4* prow4 = reinterpret_cast<const float4*>(&partS[m * T]);
            const float4* trow4 = reinterpret_cast<const float4*>(&transT[j * TTP]);
            #pragma unroll
            for (int i4 = 0; i4 < T / 4; ++i4) {
                const float4 p  = prow4[i4];
                const float4 tr = trow4[i4];
                const float m01 = fmaxf(p.x + tr.x, p.y + tr.y);
                const float m23 = fmaxf(p.z + tr.z, p.w + tr.w);
                pmax = fmaxf(pmax, fmaxf(m01, m23));
            }
        }
        const float e = eBuf[cur][tid];
        PS[tid] = pmax + 2.0f * e;
        aPerm[m * 64 + aperm_pos(j)] =
            (_Float16)__builtin_amdgcn_exp2f((partS[tid] - pmax - e) * LOG2E);
        __syncthreads();

        // Phase B:
        if (wave < 3) {
            // waves 0..2: s = a . E via two chained WMMAs (K=48 padded to 64);
            // val = (P + 2e) + ln2*log2(s); branchless masked state update.
            const int base = (lane >= 16) ? 32 : 0;
            const int am   = lane & 15;
            v16h A0 = *(const v16h*)&aPerm[am * 64 + base];
            v16h A1 = *(const v16h*)&aPerm[am * 64 + base + 16];
            v8f  c  = {};
            c = __builtin_amdgcn_wmma_f32_16x16x32_f16(false, A0, false, Bm0,
                                                       (short)0, c, false, false);
            c = __builtin_amdgcn_wmma_f32_16x16x32_f16(false, A1, false, Bm1,
                                                       (short)0, c, false, false);
            if (lane < 16) {            // C layout: lanes<16 hold M rows 0..7 (real batches)
                unsigned long long mk8;
                __builtin_memcpy(&mk8, (const void*)&maskBuf[cur][0], 8);
                const int jj = wave * 16 + lane;
                #pragma unroll
                for (int r = 0; r < 8; ++r) {
                    const int idx   = r * T + jj;
                    const float oldv = partS[idx];
                    const float val  = PS[idx] + LN2 * __builtin_amdgcn_logf(c[r]);
                    partS[idx] = ((mk8 >> (8 * r)) & 1ull) ? val : oldv;
                }
            }
        } else if (wave == 3) {
            // wave 3: TDM-stage e_{t+1} (8x48 f32 tile, row stride S*T) global -> LDS.
            if (t + 1 < S_LEN) {
                if (lane < BPB)
                    maskBuf[nxt][lane] = mask[(size_t)(b0 + lane) * S_LEN + (t + 1)];
                const uint64_t ga =
                    (uint64_t)(uintptr_t)(feats + ((size_t)b0 * S_LEN + (t + 1)) * T);
                const unsigned ldsa = (unsigned)(uintptr_t)&eBuf[nxt][0];
                // D# group0: count=1(valid), lds_addr, global_addr[56:0], type=2
                u32x4 g0 = { 1u, ldsa, (unsigned)ga,
                             (((unsigned)(ga >> 32)) & 0x01FFFFFFu) | 0x80000000u };
                // D# group1: data_size=4B; tensor_dim0=48; tensor_dim1=8;
                //            tile_dim0=48; tile_dim1=8; tensor_dim0_stride=S*T
                i32x8 g1 = { 0x00020000, (int)(48u << 16), (int)(8u << 16),
                             (int)(48u << 16), 8, (int)(S_LEN * T), 0, 0 };
                i32x4 gz4 = { 0, 0, 0, 0 };
                i32x8 gz8 = { 0, 0, 0, 0, 0, 0, 0, 0 };
                __builtin_amdgcn_tensor_load_to_lds(g0, g1, gz4, gz4, gz8, 0);
                __builtin_amdgcn_s_wait_tensorcnt(0);
            }
        } else if (wave == 4) {
            // wave 4: prefetch the t+2 emission rows into cache.
            if (t + 2 < S_LEN && lane < 2 * BPB) {
                const int mm = lane >> 1;
                const char* pf = (const char*)(feats +
                    ((size_t)(b0 + mm) * S_LEN + (t + 2)) * T) + (lane & 1) * 128;
                __builtin_prefetch(pf, 0, 1);
            }
        }
        __syncthreads();
    }

    // ---- end transition (no emission term), only column j=47 needed ----
    {
        float pmax = -3.402823466e38f;
        const float4* prow4 = reinterpret_cast<const float4*>(&partS[m * T]);
        const float4* trow4 = reinterpret_cast<const float4*>(&transT[j * TTP]);
        #pragma unroll
        for (int i4 = 0; i4 < T / 4; ++i4) {
            const float4 p  = prow4[i4];
            const float4 tr = trow4[i4];
            pmax = fmaxf(pmax, fmaxf(fmaxf(p.x + tr.x, p.y + tr.y),
                                     fmaxf(p.z + tr.z, p.w + tr.w)));
        }
        PS[tid] = pmax;     // P'[m][i]: needed for ALL i by the buggy LSE rescale
    }
    __syncthreads();
    if (tid < BPB) {
        const int mm = tid;
        float acc = 0.0f;
        for (int i = 0; i < T; ++i)
            acc += __expf(partS[mm * T + i] - PS[mm * T + i] + transT[47 * TTP + i]);
        const float score = PS[mm * T + 47] + __logf(acc);
        out[1 + b0 + mm] = score;
        scoreSh[mm] = score;
    }
    __syncthreads();
    if (tid == 0) {
        float s = 0.0f;
        for (int k = 0; k < BPB; ++k) s += scoreSh[k];
        ws[blockIdx.x] = s;     // per-block partial; deterministic final sum in kernel 2
    }
}

__global__ void crf_sum(const float* __restrict__ ws, float* __restrict__ out) {
    if (threadIdx.x == 0) {
        float s = 0.0f;
        for (int k = 0; k < NBLK; ++k) s += ws[k];
        out[0] = s;
    }
}

extern "C" void kernel_launch(void* const* d_in, const int* in_sizes, int n_in,
                              void* d_out, int out_size, void* d_ws, size_t ws_size,
                              hipStream_t stream) {
    (void)in_sizes; (void)n_in; (void)out_size; (void)ws_size;
    const float*         feats = (const float*)d_in[0];
    const unsigned char* msk   = (const unsigned char*)d_in[1];  // jnp bool_: 1 byte/elem
    const float*         trans = (const float*)d_in[2];
    float* out = (float*)d_out;
    float* ws  = (float*)d_ws;

    crf_fwd<<<NBLK, TPB, 0, stream>>>(feats, msk, trans, out, ws);
    crf_sum<<<1, 32, 0, stream>>>(ws, out);
}
